// GGE_14336600834609
// MI455X (gfx1250) — matrix operand order repaired and checked by reading, not compiled, not run on hardware
//
#include <hip/hip_runtime.h>

typedef __bf16 bf16;
typedef __attribute__((ext_vector_type(16))) __bf16 v16bf;
typedef __attribute__((ext_vector_type(8)))  float  v8f;

#define BSZ 2
#define NP  4096
#define CCH 256
#define KG  16
#define KP  32
#define RAD2 (0.15f * 0.15f)
#define EPSN 1e-5f

// ---------------------------------------------------------------------------
// Weight convert f32 -> bf16, optionally zero-padding columns (Wp1: 10 -> 32)
// ---------------------------------------------------------------------------
__global__ void k_cvt_w(const float* __restrict__ src, bf16* __restrict__ dst,
                        int rows, int cs, int cd) {
  int g = blockIdx.x * 256 + threadIdx.x;
  if (g >= rows * cd) return;
  int r = g / cd, c = g % cd;
  dst[g] = (bf16)(c < cs ? src[r * cs + c] : 0.f);
}

// feats (B,C,N) f32 -> per-point rows [b][n][C] bf16
__global__ void k_feats_pt(const float* __restrict__ f, bf16* __restrict__ out) {
  int g = blockIdx.x * 256 + threadIdx.x;
  if (g >= BSZ * CCH * NP) return;
  int n = g % NP;
  int t = g / NP;
  int c = t % CCH;
  int b = t / CCH;
  out[((size_t)(b * NP + n)) * CCH + c] = (bf16)f[g];
}

// ---------------------------------------------------------------------------
// Fused kNN(16, squared L2) + ball query (first 32 indices with d2<=r2)
// Block = 256 threads (8 waves); one wave32 per query row; coords in LDS.
// ---------------------------------------------------------------------------
__global__ void k_knn_ball(const float* __restrict__ coords,
                           int* __restrict__ knn, int* __restrict__ ball) {
  __shared__ float sx[NP], sy[NP], sz[NP];
  __shared__ int sball[8][KP];
  int b  = blockIdx.x / (NP / 8);
  int n0 = (blockIdx.x % (NP / 8)) * 8;
  const float* cb = coords + (size_t)b * 3 * NP;
  for (int i = threadIdx.x; i < NP; i += 256) {
    sx[i] = cb[i];
    sy[i] = cb[NP + i];
    sz[i] = cb[2 * NP + i];
  }
  __syncthreads();

  int wave = threadIdx.x >> 5, lane = threadIdx.x & 31;
  int n = n0 + wave;
  float px = sx[n], py = sy[n], pz = sz[n];

  float d[17];
  int   id[17];
#pragma unroll
  for (int s = 0; s < 17; ++s) { d[s] = 3e38f; id[s] = -1; }
  int cnt = 0;

  for (int j0 = 0; j0 < NP; j0 += 32) {
    int j = j0 + lane;
    float dx = sx[j] - px, dy = sy[j] - py, dz = sz[j] - pz;
    float d2 = dx * dx + dy * dy + dz * dz;
    // per-lane register top-17 (ascending)
    if (d2 < d[16]) {
      d[16] = d2; id[16] = j;
#pragma unroll
      for (int s = 16; s > 0; --s) {
        if (d[s] < d[s - 1]) {
          float td = d[s]; d[s] = d[s - 1]; d[s - 1] = td;
          int ti = id[s]; id[s] = id[s - 1]; id[s - 1] = ti;
        }
      }
    }
    // ball query in ascending index order (matches jnp.sort of indices)
    unsigned msk = (unsigned)__ballot(d2 <= RAD2);
    if (lane == 0) {
      while (msk && cnt < KP) {
        int bit = __ffs(msk) - 1;
        sball[wave][cnt++] = j0 + bit;
        msk &= msk - 1;
      }
    }
  }
  cnt = __shfl(cnt, 0, 32);
  if (lane == 0) {
    int fill = sball[wave][0];            // pad with first in-radius index
    for (int t = cnt; t < KP; ++t) sball[wave][t] = fill;
  }
  if (lane < KP) ball[((size_t)b * NP + n) * KP + lane] = sball[wave][lane];

  // extract 17 smallest across the wave; drop rank 0 (self, d=0)
#pragma unroll 1
  for (int r = 0; r < 17; ++r) {
    float bm = 3e38f; int bslot = 0;
#pragma unroll
    for (int s = 0; s < 17; ++s) if (d[s] < bm) { bm = d[s]; bslot = s; }
    int vid = -1;
#pragma unroll
    for (int s = 0; s < 17; ++s) if (s == bslot) vid = id[s];
    float v = bm; int vl = lane;
#pragma unroll
    for (int m = 16; m >= 1; m >>= 1) {
      float ov = __shfl_xor(v,   m, 32);
      int  oid = __shfl_xor(vid, m, 32);
      int  ol  = __shfl_xor(vl,  m, 32);
      if (ov < v || (ov == v && ol < vl)) { v = ov; vid = oid; vl = ol; }
    }
    if (lane == vl) {
#pragma unroll
      for (int s = 0; s < 17; ++s) if (s == bslot) d[s] = 3e38f;
    }
    if (lane == 0 && r >= 1) knn[((size_t)b * NP + n) * KG + (r - 1)] = vid;
  }
}

// ---------------------------------------------------------------------------
// PPF 10-d point-pair features, zero-padded to 32 bf16 (so GEMM K%32==0)
// ---------------------------------------------------------------------------
__global__ void k_ppf_feat(const float* __restrict__ coords,
                           const float* __restrict__ normals,
                           const int* __restrict__ ball,
                           bf16* __restrict__ pts) {
  int g = blockIdx.x * 256 + threadIdx.x;
  if (g >= BSZ * NP * KP) return;
  int n = (g / KP) % NP;
  int b = g / (KP * NP);
  const float* cx = coords  + (size_t)b * 3 * NP;
  const float* nx = normals + (size_t)b * 3 * NP;
  float px = cx[n], py = cx[NP + n], pz = cx[2 * NP + n];
  float nrx = nx[n], nry = nx[NP + n], nrz = nx[2 * NP + n];
  int j = ball[g];
  float qx = cx[j] - px, qy = cx[NP + j] - py, qz = cx[2 * NP + j] - pz;
  float nix = nx[j], niy = nx[NP + j], niz = nx[2 * NP + j];
  auto ang = [](float ax, float ay, float az, float bx, float by, float bz) {
    float cxv = ay * bz - az * by, cyv = az * bx - ax * bz, czv = ax * by - ay * bx;
    float cn = sqrtf(cxv * cxv + cyv * cyv + czv * czv + 1e-12f);
    return atan2f(cn, ax * bx + ay * by + az * bz);
  };
  float nr_d  = ang(nrx, nry, nrz, qx, qy, qz);
  float ni_d  = ang(nix, niy, niz, qx, qy, qz);
  float nr_ni = ang(nrx, nry, nrz, nix, niy, niz);
  float dn = sqrtf(qx * qx + qy * qy + qz * qz + 1e-12f);
  bf16* row = pts + (size_t)g * 32;
  row[0] = (bf16)px; row[1] = (bf16)py; row[2] = (bf16)pz;
  row[3] = (bf16)qx; row[4] = (bf16)qy; row[5] = (bf16)qz;
  row[6] = (bf16)nr_d; row[7] = (bf16)ni_d; row[8] = (bf16)nr_ni; row[9] = (bf16)dn;
#pragma unroll
  for (int i = 10; i < 32; ++i) row[i] = (bf16)0.f;
}

// ---------------------------------------------------------------------------
// Generic WMMA bf16 GEMM: out[b,o,m] = sum_c W[o,c] * X[b,m,c]
//   mode 0 DIRECT  : X = bf16 rows [b][m][Cin]
//   mode 1 GATHER  : X = per-point [b][n][Cf]; row = [f(n), f(idx)-f(n)], m=(n,k)
//   mode 2 NORMACT : X = prev pre-norm [b][m][Cin]; apply (x-mean)*rstd, (l)relu
// Accumulates per-(b,o) sum/sumsq into stats via shfl_xor + atomics.
// Block = 128 (4 waves); each wave: one 16x16 D tile, o_base = by*64+wave*16.
// ---------------------------------------------------------------------------
#define LDS_STRIDE (1024 + 8)

__global__ void k_gemm(const bf16* __restrict__ W, const bf16* __restrict__ X,
                       const int* __restrict__ idx, const float* __restrict__ ms,
                       float slope_in, float* __restrict__ stats,
                       bf16* __restrict__ Hout, float* __restrict__ Fout,
                       int Cin, int Cf, int Cout, int M, int mode) {
  __shared__ bf16 tile[16 * LDS_STRIDE];
  int b   = blockIdx.z;
  int m0  = blockIdx.x * 16;
  int tid = threadIdx.x;
  int lane = tid & 31, wave = tid >> 5;

  // ---- stage 16-sample x Cin tile into LDS (8 workers per sample) ----
  {
    int s = tid >> 3, w = tid & 7;
    int m = m0 + s;
    if (mode == 1) {
      int n = m >> 4, k = m & 15;
      const bf16* fc = X + ((size_t)(b * NP + n)) * Cf;
      int nb = idx[(b * NP + n) * KG + k];
      const bf16* fn = X + ((size_t)(b * NP + nb)) * Cf;
      for (int c = w * 8; c < Cin; c += 64) {
        bf16 o8[8];
        if (c < Cf) {
          *(uint4*)o8 = *(const uint4*)(fc + c);
        } else {
          bf16 c8[8], n8[8];
          *(uint4*)c8 = *(const uint4*)(fc + (c - Cf));
          *(uint4*)n8 = *(const uint4*)(fn + (c - Cf));
#pragma unroll
          for (int i = 0; i < 8; ++i)
            o8[i] = (bf16)((float)n8[i] - (float)c8[i]);
        }
        *(uint4*)&tile[s * LDS_STRIDE + c] = *(uint4*)o8;
      }
    } else {
      const bf16* src = X + ((size_t)b * M + m) * Cin;
      const float* msb = ms ? (ms + (size_t)b * Cin * 2) : ms;
      for (int c = w * 8; c < Cin; c += 64) {
        bf16 o8[8];
        *(uint4*)o8 = *(const uint4*)(src + c);
        if (mode == 2) {
#pragma unroll
          for (int i = 0; i < 8; ++i) {
            float f = (float)o8[i];
            f = (f - msb[2 * (c + i)]) * msb[2 * (c + i) + 1];
            f = f >= 0.f ? f : slope_in * f;
            o8[i] = (bf16)f;
          }
        }
        *(uint4*)&tile[s * LDS_STRIDE + c] = *(uint4*)o8;
      }
    }
  }
  __syncthreads();

  int o_base = blockIdx.y * 64 + wave * 16;
  int h = lane >> 4, l15 = lane & 15;
  const bf16* wrow = W + (size_t)(o_base + l15) * Cin;

  v8f acc = {};
  for (int k0 = 0; k0 < Cin; k0 += 32) {
    union { v16bf v; uint4 q[2]; } a, bm;
    // A (16x32 bf16): lanes0-15 rows M=l15, K {k0..k0+7, k0+16..k0+23};
    // lanes16-31 same rows, K shifted by 8 (ISA 7.12.2 layout)
    a.q[0] = *(const uint4*)(wrow + k0 + 8 * h);
    a.q[1] = *(const uint4*)(wrow + k0 + 16 + 8 * h);
    if (k0 + 32 < Cin) __builtin_prefetch(wrow + k0 + 32, 0, 1);
    // B (32x16 bf16): lane -> column n=l15; K same as A {8h..8h+7, 16+8h..16+8h+7}
    const bf16* brow = &tile[l15 * LDS_STRIDE + k0 + 8 * h];
    bm.q[0] = *(const uint4*)(brow);
    bm.q[1] = *(const uint4*)(brow + 16);
    acc = __builtin_amdgcn_wmma_f32_16x16x32_bf16(false, a.v, false, bm.v,
                                                  (short)0, acc, false, false);
  }

  // ---- per-(b,o) sum / sumsq for instance norm (reduce 16 m-columns) ----
#pragma unroll
  for (int j = 0; j < 8; ++j) {
    float v = acc[j];
    float s = v, q = v * v;
#pragma unroll
    for (int m = 1; m <= 8; m <<= 1) {
      s += __shfl_xor(s, m, 32);
      q += __shfl_xor(q, m, 32);
    }
    if (l15 == 0) {
      int o = o_base + 8 * h + j;
      atomicAdd(&stats[((size_t)b * Cout + o) * 2],     s);
      atomicAdd(&stats[((size_t)b * Cout + o) * 2 + 1], q);
    }
  }

  // ---- store pre-norm result ----
  int mcol = m0 + l15;
  if (Fout) {  // f32 [b][o][m]
#pragma unroll
    for (int j = 0; j < 8; ++j)
      Fout[((size_t)b * Cout + (o_base + 8 * h + j)) * M + mcol] = acc[j];
  } else {     // bf16 [b][m][o] : lane's 8 o-values are contiguous -> 16B store
    bf16 o8[8];
#pragma unroll
    for (int j = 0; j < 8; ++j) o8[j] = (bf16)acc[j];
    *(uint4*)&Hout[((size_t)b * M + mcol) * Cout + o_base + 8 * h] = *(uint4*)o8;
  }
}

// stats (sum,sumsq) -> (mean, rstd), in place; count = B*Cout entries
__global__ void k_finalize(float* __restrict__ stats, int count, float invM) {
  int g = blockIdx.x * 256 + threadIdx.x;
  if (g >= count) return;
  float mean = stats[2 * g] * invM;
  float var  = stats[2 * g + 1] * invM - mean * mean;
  stats[2 * g]     = mean;
  stats[2 * g + 1] = rsqrtf(var + EPSN);
}

// out_pt[b][n][o] = max_k act((H[b][n*K+k][o]-mean)*rstd)
__global__ void k_reduce_max(const bf16* __restrict__ H, const float* __restrict__ ms,
                             bf16* __restrict__ outPt, int Cout, int K, float slope) {
  size_t g = (size_t)blockIdx.x * 256 + threadIdx.x;
  if (g >= (size_t)BSZ * NP * Cout) return;
  int o = g % Cout;
  size_t bn = g / Cout;
  int n = bn % NP, b = bn / NP;
  float mean = ms[((size_t)b * Cout + o) * 2];
  float rs   = ms[((size_t)b * Cout + o) * 2 + 1];
  float best = -3e38f;
  const bf16* base = H + ((size_t)b * NP * K + (size_t)n * K) * Cout + o;
  for (int k = 0; k < K; ++k) {
    float v = (float)base[(size_t)k * Cout];
    v = (v - mean) * rs;
    v = v >= 0.f ? v : slope * v;
    best = best > v ? best : v;
  }
  outPt[((size_t)b * NP + n) * Cout + o] = (bf16)best;
}

// x3[b][n][1024] = [feats(256) | f1(256) | f2(512)]
__global__ void k_concat3(const bf16* __restrict__ fp, const bf16* __restrict__ f1,
                          const bf16* __restrict__ f2, bf16* __restrict__ x3) {
  size_t g = (size_t)blockIdx.x * 256 + threadIdx.x;
  if (g >= (size_t)BSZ * NP * 1024) return;
  int c = g % 1024;
  size_t bn = g / 1024;
  bf16 v;
  if (c < 256)      v = fp[bn * 256 + c];
  else if (c < 512) v = f1[bn * 256 + (c - 256)];
  else              v = f2[bn * 512 + (c - 512)];
  x3[g] = v;
}

// xh[b][n][384] = [ppf_out(128) | lrelu(norm(h3))(256)]
__global__ void k_head_concat(const bf16* __restrict__ ppfOut,
                              const float* __restrict__ h3,
                              const float* __restrict__ ms3,
                              bf16* __restrict__ xh) {
  size_t g = (size_t)blockIdx.x * 256 + threadIdx.x;
  if (g >= (size_t)BSZ * NP * 384) return;
  int c = g % 384;
  size_t bn = g / 384;
  int n = bn % NP, b = bn / NP;
  bf16 v;
  if (c < 128) {
    v = ppfOut[bn * 128 + c];
  } else {
    int o = c - 128;
    float f = h3[((size_t)b * 256 + o) * NP + n];
    f = (f - ms3[((size_t)b * 256 + o) * 2]) * ms3[((size_t)b * 256 + o) * 2 + 1];
    f = f >= 0.f ? f : 0.2f * f;
    v = (bf16)f;
  }
  xh[g] = v;
}

// d_out[b][o][n] = lrelu((hf - mean) * rstd)
__global__ void k_final(const float* __restrict__ hf, const float* __restrict__ ms,
                        float* __restrict__ out) {
  size_t g = (size_t)blockIdx.x * 256 + threadIdx.x;
  if (g >= (size_t)BSZ * 256 * NP) return;
  size_t bo = g / NP;
  float f = hf[g];
  f = (f - ms[bo * 2]) * ms[bo * 2 + 1];
  out[g] = f >= 0.f ? f : 0.2f * f;
}

// ---------------------------------------------------------------------------
extern "C" void kernel_launch(void* const* d_in, const int* in_sizes, int n_in,
                              void* d_out, int out_size, void* d_ws, size_t ws_size,
                              hipStream_t stream) {
  (void)in_sizes; (void)n_in; (void)out_size; (void)ws_size;
  const float* coords  = (const float*)d_in[0];
  const float* feats   = (const float*)d_in[1];
  const float* normals = (const float*)d_in[2];
  const float* W1 = (const float*)d_in[3];
  const float* W2 = (const float*)d_in[4];
  const float* W3 = (const float*)d_in[5];
  const float* Wp1 = (const float*)d_in[6];
  const float* Wp2 = (const float*)d_in[7];
  const float* Wp3 = (const float*)d_in[8];
  const float* Wf1 = (const float*)d_in[9];
  // d_in[10] = bf1, d_in[12] = bf2: constant bias before instance-norm is a
  // no-op (norm subtracts the mean), so they are intentionally unused.
  const float* Wf2 = (const float*)d_in[11];
  float* out = (float*)d_out;

  // ---- workspace carve (256B aligned) ----
  char* p = (char*)d_ws;
  auto carve = [&](size_t bytes) {
    void* r = (void*)p;
    p += (bytes + 255) & ~(size_t)255;
    return r;
  };
  bf16* wW1  = (bf16*)carve(256 * 512 * 2);
  bf16* wW2  = (bf16*)carve(512 * 512 * 2);
  bf16* wW3  = (bf16*)carve(256 * 1024 * 2);
  bf16* wWp1 = (bf16*)carve(128 * 32 * 2);
  bf16* wWp2 = (bf16*)carve(256 * 128 * 2);
  bf16* wWp3 = (bf16*)carve(128 * 256 * 2);
  bf16* wWf1 = (bf16*)carve(384 * 384 * 2);
  bf16* wWf2 = (bf16*)carve(256 * 384 * 2);
  int*  knn  = (int*)carve((size_t)BSZ * NP * KG * 4);
  int*  ball = (int*)carve((size_t)BSZ * NP * KP * 4);
  bf16* featsPt = (bf16*)carve((size_t)BSZ * NP * 256 * 2);
  bf16* ppfPts  = (bf16*)carve((size_t)BSZ * NP * KP * 32 * 2);
  bf16* f1Pt    = (bf16*)carve((size_t)BSZ * NP * 256 * 2);
  bf16* f2Pt    = (bf16*)carve((size_t)BSZ * NP * 512 * 2);
  bf16* ppfOut  = (bf16*)carve((size_t)BSZ * NP * 128 * 2);
  bf16* x3      = (bf16*)carve((size_t)BSZ * NP * 1024 * 2);
  bf16* xh      = (bf16*)carve((size_t)BSZ * NP * 384 * 2);
  float* h3 = (float*)carve((size_t)BSZ * 256 * NP * 4);
  float* hf = (float*)carve((size_t)BSZ * 256 * NP * 4);
  float* stats = (float*)carve(8 * 4096 * 4);             // 8 slots x 4096 f32
  bf16* HA = (bf16*)carve((size_t)BSZ * 65536 * 512 * 2); // 134 MB ping
  bf16* HB = (bf16*)carve((size_t)BSZ * 131072 * 256 * 2);// 134 MB pong
  auto slot = [&](int i) { return stats + (size_t)i * 4096; };

  hipMemsetAsync(stats, 0, 8 * 4096 * 4, stream);

  // ---- weights -> bf16 ----
  auto cvt = [&](const float* s, bf16* d, int r, int cs, int cd) {
    k_cvt_w<<<(r * cd + 255) / 256, 256, 0, stream>>>(s, d, r, cs, cd);
  };
  cvt(W1, wW1, 256, 512, 512);
  cvt(W2, wW2, 512, 512, 512);
  cvt(W3, wW3, 256, 1024, 1024);
  cvt(Wp1, wWp1, 128, 10, 32);
  cvt(Wp2, wWp2, 256, 128, 128);
  cvt(Wp3, wWp3, 128, 256, 256);
  cvt(Wf1, wWf1, 384, 384, 384);
  cvt(Wf2, wWf2, 256, 384, 384);

  k_feats_pt<<<(BSZ * CCH * NP + 255) / 256, 256, 0, stream>>>(feats, featsPt);
  k_knn_ball<<<BSZ * (NP / 8), 256, 0, stream>>>(coords, knn, ball);
  k_ppf_feat<<<(BSZ * NP * KP + 255) / 256, 256, 0, stream>>>(coords, normals, ball, ppfPts);

  auto gemm = [&](const bf16* W, const bf16* X, const int* idx, const float* ms,
                  float slope, float* st, bf16* Hout, float* Fout,
                  int Cin, int Cf, int Cout, int M, int mode) {
    dim3 g(M / 16, Cout / 64, BSZ);
    k_gemm<<<g, 128, 0, stream>>>(W, X, idx, ms, slope, st, Hout, Fout,
                                  Cin, Cf, Cout, M, mode);
  };
  auto fin = [&](float* st, int Cout, int M) {
    k_finalize<<<(BSZ * Cout + 255) / 256, 256, 0, stream>>>(st, BSZ * Cout, 1.f / (float)M);
  };
  auto rmax = [&](const bf16* H, const float* ms, bf16* o, int Cout, int K, float sl) {
    k_reduce_max<<<(int)(((size_t)BSZ * NP * Cout + 255) / 256), 256, 0, stream>>>(H, ms, o, Cout, K, sl);
  };

  // ---- GCN branch ----
  gemm(wW1, featsPt, knn, nullptr, 0.f, slot(0), HA, nullptr, 512, 256, 256, NP * KG, 1);
  fin(slot(0), 256, NP * KG);
  rmax(HA, slot(0), f1Pt, 256, KG, 0.2f);

  gemm(wW2, f1Pt, knn, nullptr, 0.f, slot(1), HA, nullptr, 512, 256, 512, NP * KG, 1);
  fin(slot(1), 512, NP * KG);
  rmax(HA, slot(1), f2Pt, 512, KG, 0.2f);

  k_concat3<<<(int)(((size_t)BSZ * NP * 1024 + 255) / 256), 256, 0, stream>>>(featsPt, f1Pt, f2Pt, x3);
  gemm(wW3, x3, nullptr, nullptr, 0.f, slot(2), nullptr, h3, 1024, 0, 256, NP, 0);
  fin(slot(2), 256, NP);

  // ---- PPF branch ----
  gemm(wWp1, ppfPts, nullptr, nullptr, 0.f, slot(3), HA, nullptr, 32, 0, 128, NP * KP, 0);
  fin(slot(3), 128, NP * KP);
  gemm(wWp2, HA, nullptr, slot(3), 0.f, slot(4), HB, nullptr, 128, 0, 256, NP * KP, 2);
  fin(slot(4), 256, NP * KP);
  gemm(wWp3, HB, nullptr, slot(4), 0.f, slot(5), HA, nullptr, 256, 0, 128, NP * KP, 2);
  fin(slot(5), 128, NP * KP);
  rmax(HA, slot(5), ppfOut, 128, KP, 0.f);

  // ---- head ----
  k_head_concat<<<(int)(((size_t)BSZ * NP * 384 + 255) / 256), 256, 0, stream>>>(ppfOut, h3, slot(2), xh);
  gemm(wWf1, xh, nullptr, nullptr, 0.f, slot(6), HA, nullptr, 384, 0, 384, NP, 0);
  fin(slot(6), 384, NP);
  gemm(wWf2, HA, nullptr, slot(6), 0.2f, slot(7), nullptr, hf, 384, 0, 256, NP, 2);
  fin(slot(7), 256, NP);
  k_final<<<(int)(((size_t)BSZ * 256 * NP + 255) / 256), 256, 0, stream>>>(hf, slot(7), out);
}
